// LearnedConservativeSolver_41807211659834
// MI455X (gfx1250) — matrix operand order repaired and checked by reading, not compiled, not run on hardware
//
#include <hip/hip_runtime.h>

typedef _Float16 f16;
typedef __attribute__((ext_vector_type(16))) _Float16 v16h;
typedef __attribute__((ext_vector_type(8)))  _Float16 v8h;
typedef __attribute__((ext_vector_type(2)))  _Float16 v2h;
typedef __attribute__((ext_vector_type(8)))  float    v8f;

#define B_DIM 64
#define N_DIM 4096
#define NF    (N_DIM + 1)   // 4097 interfaces per row
#define H     64
#define NTILES 257          // ceil(4097 / 16)

// tanh(x) = 2/(1+exp(-2x)) - 1 = 2*rcp(1 + exp2(SC*x)) - 1, SC = -2*log2(e).
// We pre-scale weights/biases by SC so the network directly produces z = SC*x.
#define TANH_SCALE (-2.8853900817779268f)   // -2 * log2(e)

union V16Pack { v16h v16; v8h v8[2]; };

// Input z is already scaled by TANH_SCALE. Saturates cleanly (no NaN path):
// exp2->inf => rcp->0 => -1 ; exp2->0 => rcp(1)=1 => +1.
__device__ __forceinline__ float tanh_from_z(float z) {
  float t = __builtin_amdgcn_exp2f(z);
  float r = __builtin_amdgcn_rcpf(1.0f + t);
  return fmaf(2.0f, r, -1.0f);
}

// A-operand loader: contiguous k order shared with B (lanes<16: K 0-15; lanes>=16: K 16-31)
__device__ __forceinline__ v16h load_A(const f16* base, int hi) {
  V16Pack u;
  const f16* p = base + (hi ? 16 : 0);
  u.v8[0] = *(const v8h*)(p);
  u.v8[1] = *(const v8h*)(p + 8);
  return u.v16;
}

__global__ __launch_bounds__(256) void flux_kernel(
    const float* __restrict__ u,    // [B, N] current state
    const float* __restrict__ W1,   // [2, 64]
    const float* __restrict__ b1,   // [64]
    const float* __restrict__ W2,   // [64, 64] (k-major)
    const float* __restrict__ b2,   // [64]
    const float* __restrict__ W3,   // [64]
    const float* __restrict__ b3,   // [1]
    float* __restrict__ fbuf)       // [B, N+1]
{
  __shared__ f16 sW2T[H * H];        // (TANH_SCALE * W2)^T : [n][k], f16
  __shared__ f16 sH1[8][16 * H];     // per-wave h1 tile: [16 rows][64], f16

  const int tid  = threadIdx.x;
  const int wave = tid >> 5;
  const int lane = tid & 31;
  const int ln   = lane & 15;
  const int hi   = lane >> 4;        // half-wave id
  const int t    = blockIdx.x;       // interface tile column

  // ---- stage SC*W2 transposed into LDS (f16) ----
  for (int e = tid; e < H * H; e += 256) {
    int k = e >> 6, n = e & 63;
    sW2T[n * H + k] = (f16)(W2[e] * TANH_SCALE);
  }
  __syncthreads();

  // ---- per-lane constants (layer-1 weights pre-scaled by SC) ----
  const int   j0  = 2 * lane, j1 = 2 * lane + 1;
  const float wA0 = W1[j0]      * TANH_SCALE, wA1 = W1[j1]      * TANH_SCALE;
  const float wB0 = W1[64 + j0] * TANH_SCALE, wB1 = W1[64 + j1] * TANH_SCALE;
  const float bb0 = b1[j0]      * TANH_SCALE, bb1 = b1[j1]      * TANH_SCALE;
  float b2v[4], w3v[4];
#pragma unroll
  for (int q = 0; q < 4; q++) {
    b2v[q] = b2[16 * q + ln] * TANH_SCALE;   // accumulator is SC*(h1@W2); add SC*b2
    w3v[q] = W3[16 * q + ln];
  }
  const float b3s = b3[0];

  // ---- B operands (SC*W2 tiles), register-resident across the row loop ----
  // B k order used here (same as A): N = lane%16; lanes<16 hold K 0-15, lanes>=16 hold K 16-31 (contiguous)
  v16h Bm[4][2];
#pragma unroll
  for (int q = 0; q < 4; q++)
#pragma unroll
    for (int k = 0; k < 2; k++) {
      const f16* p = &sW2T[(16 * q + ln) * H + 32 * k + (hi ? 16 : 0)];
      Bm[q][k] = *(const v16h*)p;
    }

  // interface index handled by this lane's row slot (m = ln)
  const int i  = 16 * t + ln;
  const int iL = min(max(i - 1, 0), N_DIM - 1);
  const int iR = min(i, N_DIM - 1);

  f16* h1 = &sH1[wave][0];

  for (int b = wave; b < B_DIM; b += 8) {
    const float* urow = u + (size_t)b * N_DIM;
    const float uL = urow[iL];
    const float uR = urow[iR];

    // ---- layer 1: h1[16][64] = tanh(x @ W1 + b1), row p broadcast via shfl ----
#pragma unroll
    for (int p = 0; p < 16; p++) {
      float ul = __shfl(uL, p, 32);
      float ur = __shfl(uR, p, 32);
      float a0 = tanh_from_z(fmaf(ul, wA0, fmaf(ur, wB0, bb0)));
      float a1 = tanh_from_z(fmaf(ul, wA1, fmaf(ur, wB1, bb1)));
      v2h pk = { (f16)a0, (f16)a1 };
      *(v2h*)&h1[p * H + j0] = pk;
    }
    __builtin_amdgcn_wave_barrier();   // stores above precede loads below (DS is in-order per wave)

    // ---- layer 2: SC*(h1 @ W2) via 8x v_wmma_f32_16x16x32_f16 ----
    const f16* h1r = &h1[ln * H];
    v16h A0 = load_A(h1r,      hi);    // K 0..31
    v16h A1 = load_A(h1r + 32, hi);    // K 32..63

    float prt[8] = {0.f, 0.f, 0.f, 0.f, 0.f, 0.f, 0.f, 0.f};
#pragma unroll
    for (int q = 0; q < 4; q++) {
      v8f c = {};
      c = __builtin_amdgcn_wmma_f32_16x16x32_f16(false, A0, false, Bm[q][0],
                                                 (short)0, c, false, false);
      c = __builtin_amdgcn_wmma_f32_16x16x32_f16(false, A1, false, Bm[q][1],
                                                 (short)0, c, false, false);
      // layer 3 partials: tanh(h2) * W3, per C-layout (M = r + 8*hi, N = 16q + ln)
#pragma unroll
      for (int r = 0; r < 8; r++) {
        float h2 = tanh_from_z(c[r] + b2v[q]);
        prt[r] = fmaf(h2, w3v[q], prt[r]);
      }
    }
    __builtin_amdgcn_wave_barrier();   // keep next iteration's h1 stores after these loads

    // ---- reduce over N within each 16-lane half; add b3 ----
#pragma unroll
    for (int r = 0; r < 8; r++) {
      float s = prt[r];
      s += __shfl_xor(s, 1, 32);
      s += __shfl_xor(s, 2, 32);
      s += __shfl_xor(s, 4, 32);
      s += __shfl_xor(s, 8, 32);
      prt[r] = s + b3s;
    }

    // lane 0 writes rows 0-7, lane 16 writes rows 8-15
    if (ln == 0) {
      const int mbase = 16 * t + 8 * hi;
      float* frow = fbuf + (size_t)b * NF;
#pragma unroll
      for (int r = 0; r < 8; r++) {
        int ii = mbase + r;
        if (ii < NF) frow[ii] = prt[r];
      }
    }
  }
}

__global__ __launch_bounds__(256) void update_kernel(
    const float* __restrict__ u,      // [B, N] state s
    const float* __restrict__ fbuf,   // [B, N+1]
    const float* __restrict__ dt,
    const float* __restrict__ dx,
    float* __restrict__ unew)         // [B, N] state s+1
{
  int idx = blockIdx.x * blockDim.x + threadIdx.x;
  if (idx >= B_DIM * N_DIM) return;
  int b = idx >> 12;
  int i = idx & (N_DIM - 1);
  float coeff = dt[0] * __builtin_amdgcn_rcpf(dx[0]);
  const float* fr = fbuf + (size_t)b * NF;
  unew[idx] = u[idx] - coeff * (fr[i + 1] - fr[i]);
}

extern "C" void kernel_launch(void* const* d_in, const int* in_sizes, int n_in,
                              void* d_out, int out_size, void* d_ws, size_t ws_size,
                              hipStream_t stream) {
  const float* u0 = (const float*)d_in[0];
  const float* W1 = (const float*)d_in[1];
  const float* b1 = (const float*)d_in[2];
  const float* W2 = (const float*)d_in[3];
  const float* b2 = (const float*)d_in[4];
  const float* W3 = (const float*)d_in[5];
  const float* b3 = (const float*)d_in[6];
  const float* dt = (const float*)d_in[7];
  const float* dx = (const float*)d_in[8];
  (void)n_in; (void)in_sizes; (void)ws_size;

  const int BN = B_DIM * N_DIM;
  const int nsteps = out_size / BN - 1;   // trajectory is [nsteps+1, B, N]

  float* out  = (float*)d_out;
  float* fbuf = (float*)d_ws;             // needs B*(N+1)*4 ≈ 1.05 MB scratch

  // trajectory[0] = u0
  hipMemcpyAsync(out, u0, (size_t)BN * sizeof(float),
                 hipMemcpyDeviceToDevice, stream);

  for (int s = 0; s < nsteps; s++) {
    const float* ucur = out + (size_t)s * BN;
    float*       unxt = out + (size_t)(s + 1) * BN;
    flux_kernel<<<NTILES, 256, 0, stream>>>(ucur, W1, b1, W2, b2, W3, b3, fbuf);
    update_kernel<<<(BN + 255) / 256, 256, 0, stream>>>(ucur, fbuf, dt, dx, unxt);
  }
}
